// LayoutLMv3SelfAttention_49873160241678
// MI455X (gfx1250) — matrix operand order repaired and checked by reading, not compiled, not run on hardware
//
#include <hip/hip_runtime.h>
#include <hip/hip_bf16.h>

// Problem constants (LayoutLMv3 self-attention, eval mode)
#define B_   4
#define S_   1024
#define HID_ 768
#define NH_  12
#define HD_  64
#define NQKV (3 * HID_)          // 2304 packed output columns (q|k|v)

#define SC_BYTES  (16 * S_ * sizeof(float))      // 64 KB f32 scores
#define PF_BYTES  (16 * S_ * sizeof(_Float16))   // 32 KB f16 probs
#define REL_OFF   (SC_BYTES + PF_BYTES)          // rel-bias staging offset
#define REL_BYTES (2 * 2 * 16 * 16 * sizeof(float)) // 2 parity x 2 tensors x 1KB

typedef __attribute__((ext_vector_type(16))) _Float16 v16h;
typedef __attribute__((ext_vector_type(8)))  _Float16 v8h;
typedef __attribute__((ext_vector_type(8)))  float    v8f;

// Merge two contiguous 16-byte chunks into a 16x32 f16 A-fragment register set.
#define MERGE16(lo, hi) __builtin_shufflevector((lo), (hi), 0, 1, 2, 3, 4, 5, 6, 7, \
                                                8, 9, 10, 11, 12, 13, 14, 15)

// Async copy 16B global -> LDS, tracked by ASYNCcnt (CDNA5 §10.7 / §15.18.3).
__device__ __forceinline__ void async_b128_to_lds(const float* gsrc, unsigned lds_dst) {
    const unsigned long long ga = (unsigned long long)(uintptr_t)gsrc;
    asm volatile("global_load_async_to_lds_b128 %0, %1, off"
                 :: "v"(lds_dst), "v"(ga)
                 : "memory");
}

// ---------------------------------------------------------------------------
// Pack / convert kernels (f32 -> f16 staging in workspace)
// ---------------------------------------------------------------------------
__global__ void cvt_x_kernel(const float* __restrict__ x, _Float16* __restrict__ x16, int n) {
    int i = blockIdx.x * blockDim.x + threadIdx.x;
    if (i < n) x16[i] = (_Float16)x[i];
}

// Pack W transposed: W16t[n][k] = W_which[k][c]  (n over 2304, k over 768)
// so B fragments (per-lane fixed col n, contiguous k) are single v16h loads.
__global__ void pack_w_kernel(const float* __restrict__ Wq, const float* __restrict__ Wk,
                              const float* __restrict__ Wv, _Float16* __restrict__ W16t) {
    int i = blockIdx.x * blockDim.x + threadIdx.x;        // over [NQKV][HID_]
    if (i >= NQKV * HID_) return;
    int n = i / HID_;
    int k = i % HID_;
    int which = n / HID_;
    int c = n % HID_;
    const float* Wsrc = (which == 0) ? Wq : (which == 1 ? Wk : Wv);
    W16t[i] = (_Float16)Wsrc[k * HID_ + c];
}

__global__ void pack_bias_kernel(const float* __restrict__ bq, const float* __restrict__ bk,
                                 const float* __restrict__ bv, float* __restrict__ bqkv) {
    int i = blockIdx.x * blockDim.x + threadIdx.x;
    if (i >= NQKV) return;
    int which = i / HID_;
    int c = i % HID_;
    bqkv[i] = (which == 0) ? bq[c] : (which == 1 ? bk[c] : bv[c]);
}

// ---------------------------------------------------------------------------
// Fused QKV projection GEMM: X16[4096x768] @ W16t^T + bias.
// One 16x16 output tile per wave; 24 K-steps of v_wmma_f32_16x16x32_f16.
// Q,K scatter to [B,NH,S,HD]; V scatters TRANSPOSED to [B,NH,HD,S].
// Q pre-scaled by 1/sqrt(d).
// ---------------------------------------------------------------------------
__global__ __launch_bounds__(128)
void qkv_gemm_kernel(const _Float16* __restrict__ X, const _Float16* __restrict__ Wt,
                     const float* __restrict__ bias,
                     _Float16* __restrict__ Q, _Float16* __restrict__ K,
                     _Float16* __restrict__ VT) {
    const int wave = threadIdx.x >> 5;
    const int lane = threadIdx.x & 31;
    const int half = lane >> 4;
    const int lid  = lane & 15;
    const int m0 = blockIdx.y * 16;                       // row tile (B*S)
    const int n0 = (blockIdx.x * 4 + wave) * 16;          // col tile (3*HID)
    const float inv_sqrt_d = 0.125f;                      // 1/sqrt(64)

    const _Float16* xrow = X  + (size_t)(m0 + lid) * HID_ + half * 16;
    const _Float16* wrow = Wt + (size_t)(n0 + lid) * HID_ + half * 16;

    v8f acc = {};
#pragma unroll 4
    for (int kt = 0; kt < HID_ / 32; ++kt) {
        const int kb = kt * 32;
        // A fragment: two b128 loads (K = kb + half*16 + e, and +8)
        const v8h a_lo = *(const v8h*)(xrow + kb);
        const v8h a_hi = *(const v8h*)(xrow + kb + 8);
        const v16h a = MERGE16(a_lo, a_hi);
        // B fragment: 16 contiguous halves (K = kb + half*16 + e), 32B aligned
        const v16h b = *(const v16h*)(wrow + kb);
        acc = __builtin_amdgcn_wmma_f32_16x16x32_f16(false, a, false, b,
                                                     (short)0, acc, false, false);
    }

#pragma unroll
    for (int r = 0; r < 8; ++r) {
        const int m = m0 + half * 8 + r;
        const int n = n0 + lid;
        float v = acc[r] + bias[n];
        const int which = n / HID_;
        const int c = n % HID_;
        const int h = c >> 6, d = c & 63;
        const int bb = m >> 10, s = m & 1023;
        if (which == 0) {
            Q[(((size_t)bb * NH_ + h) * S_ + s) * HD_ + d] = (_Float16)(v * inv_sqrt_d);
        } else if (which == 1) {
            K[(((size_t)bb * NH_ + h) * S_ + s) * HD_ + d] = (_Float16)v;
        } else {
            // transposed: [b,h,d,s] -> key-contiguous for P@V B fragments
            VT[(((size_t)bb * NH_ + h) * HD_ + d) * S_ + s] = (_Float16)v;
        }
    }
}

// ---------------------------------------------------------------------------
// Attention: one wave handles 16 query rows of one (b,h).
// Phase 1: scores (Q @ K^T) + (rel+rel2)/sqrt(d) + mask  -> LDS f32 [16 x 1024]
//          rel-bias tiles double-buffered through LDS via async b128 copies
// Phase 2: PB-relax softmax == standard safe softmax; probs -> LDS f16 buffer
// Phase 3: ctx = P @ V^T, normalize, write [B,S,HID] f32
// LDS: 64 KB scores + 32 KB probs + 4 KB bias staging = 100 KB dynamic.
// ---------------------------------------------------------------------------
__global__ __launch_bounds__(32)
void attn_kernel(const _Float16* __restrict__ Q, const _Float16* __restrict__ Km,
                 const _Float16* __restrict__ VT,
                 const float* __restrict__ relP, const float* __restrict__ rel2,
                 const float* __restrict__ mask, float* __restrict__ out) {
    extern __shared__ char smem[];
    float*    sc  = (float*)smem;                          // [16][1024] f32
    _Float16* pf  = (_Float16*)(smem + SC_BYTES);          // [16][1024] f16
    float*    rb  = (float*)(smem + REL_OFF);              // [2 parity][2 tensors][256] f32
    const unsigned rb_lds = (unsigned)(uintptr_t)rb;       // LDS byte address (addr[31:0])

    const int lane = threadIdx.x & 31;
    const int half = lane >> 4;
    const int lid  = lane & 15;
    const int q0 = blockIdx.x * 16;
    const int bh = blockIdx.y;
    const int b  = bh / NH_;
    const int h  = bh % NH_;
    const float inv_sqrt_d = 0.125f;

    const _Float16* Qb = Q  + (size_t)bh * S_ * HD_;
    const _Float16* Kb = Km + (size_t)bh * S_ * HD_;
    const _Float16* Vb = VT + (size_t)bh * HD_ * S_;       // [d][s]
    const float* rP = relP + (size_t)bh * S_ * S_;
    const float* r2 = rel2 + (size_t)bh * S_ * S_;
    const float* mk = mask + (size_t)b * S_;

    // Per-lane b128 chunk ids of a 16x16 f32 tile (64 chunks, 2 per lane).
    const int c0 = lane, c1 = lane + 32;

    // Issue the 4 async b128 copies staging tile `nt` into parity buffer `par`.
    auto stage_rel_tile = [&](int nt, int par) {
        const int nn = nt * 16;
        const unsigned base = rb_lds + (unsigned)par * 2048u;
        async_b128_to_lds(rP + (size_t)(q0 + (c0 >> 2)) * S_ + nn + (c0 & 3) * 4,
                          base + (unsigned)c0 * 16u);
        async_b128_to_lds(rP + (size_t)(q0 + (c1 >> 2)) * S_ + nn + (c1 & 3) * 4,
                          base + (unsigned)c1 * 16u);
        async_b128_to_lds(r2 + (size_t)(q0 + (c0 >> 2)) * S_ + nn + (c0 & 3) * 4,
                          base + 1024u + (unsigned)c0 * 16u);
        async_b128_to_lds(r2 + (size_t)(q0 + (c1 >> 2)) * S_ + nn + (c1 & 3) * 4,
                          base + 1024u + (unsigned)c1 * 16u);
    };

    // Q row fragments are loop-invariant over key tiles: preload both hd chunks.
    const _Float16* qrow = Qb + (size_t)(q0 + lid) * HD_ + half * 16;
    const v16h aq0 = MERGE16(*(const v8h*)(qrow + 0),  *(const v8h*)(qrow + 8));
    const v16h aq1 = MERGE16(*(const v8h*)(qrow + 32), *(const v8h*)(qrow + 40));

    // ---- Phase 1: score tiles (rel biases pipelined through LDS) ----
    stage_rel_tile(0, 0);                                  // prefetch tile 0
    for (int nt = 0; nt < S_ / 16; ++nt) {
        const int n0 = nt * 16;
        const int par = nt & 1;
        if (nt + 1 < S_ / 16) {
            stage_rel_tile(nt + 1, par ^ 1);               // prefetch next tile
            // async loads retire in order: <=4 outstanding => tile nt resident
            asm volatile("s_wait_asynccnt 0x4" ::: "memory");
        } else {
            asm volatile("s_wait_asynccnt 0x0" ::: "memory");
        }

        const _Float16* krow = Kb + (size_t)(n0 + lid) * HD_ + half * 16;
        v8f acc = {};
        {   // hd chunk 0 (K = 0..31)
            const v16h bk0 = *(const v16h*)(krow + 0);
            acc = __builtin_amdgcn_wmma_f32_16x16x32_f16(false, aq0, false, bk0,
                                                         (short)0, acc, false, false);
        }
        {   // hd chunk 1 (K = 32..63)
            const v16h bk1 = *(const v16h*)(krow + 32);
            acc = __builtin_amdgcn_wmma_f32_16x16x32_f16(false, aq1, false, bk1,
                                                         (short)0, acc, false, false);
        }

        const float* rt = rb + par * 512;                  // [tensor0 256][tensor1 256]
        const float mval = mk[n0 + lid];
#pragma unroll
        for (int r = 0; r < 8; ++r) {
            const int m = half * 8 + r;
            const float bias2 = (rt[m * 16 + lid] + rt[256 + m * 16 + lid]) * inv_sqrt_d;
            sc[m * S_ + n0 + lid] = acc[r] + bias2 + mval;
        }
    }
    __syncthreads();

    // ---- Phase 2: safe softmax (== PB-relax: (x/a - max(x/a))*a = x - max(x)) ----
    const int rowm = lane >> 1;                            // 2 lanes per row
    const int seg  = lane & 1;
    float rmax = -3.0e38f;
    for (int j = seg * (S_ / 2); j < (seg + 1) * (S_ / 2); ++j)
        rmax = fmaxf(rmax, sc[rowm * S_ + j]);
    rmax = fmaxf(rmax, __shfl_xor(rmax, 1));
    float rsum = 0.f;
    for (int j = seg * (S_ / 2); j < (seg + 1) * (S_ / 2); ++j) {
        const float e = __expf(sc[rowm * S_ + j] - rmax);
        pf[rowm * S_ + j] = (_Float16)e;                   // f16 probs buffer
        rsum += e;
    }
    rsum += __shfl_xor(rsum, 1);
    __syncthreads();

    // ---- Phase 3: ctx = P @ V^T ----
    v8f o[4] = {v8f{}, v8f{}, v8f{}, v8f{}};
    const _Float16* prow = pf + (size_t)lid * S_ + half * 16;
    for (int kb = 0; kb < S_; kb += 32) {
        // A fragment from LDS probs: two ds_load_b128
        const v16h a = MERGE16(*(const v8h*)(prow + kb), *(const v8h*)(prow + kb + 8));
#pragma unroll
        for (int dt = 0; dt < 4; ++dt) {                   // HD = 4 x 16 cols
            const int d = dt * 16 + lid;
            // B fragment from transposed V: 16 contiguous keys, one v16h
            const v16h bv = *(const v16h*)(Vb + (size_t)d * S_ + kb + half * 16);
            o[dt] = __builtin_amdgcn_wmma_f32_16x16x32_f16(false, a, false, bv,
                                                           (short)0, o[dt], false, false);
        }
    }

#pragma unroll
    for (int r = 0; r < 8; ++r) {
        const int m = half * 8 + r;
        const float inv = 1.0f / __shfl(rsum, 2 * m);      // row sum of row m
#pragma unroll
        for (int dt = 0; dt < 4; ++dt) {
            const int d = dt * 16 + lid;
            out[((size_t)b * S_ + (q0 + m)) * HID_ + h * HD_ + d] = o[dt][r] * inv;
        }
    }
}

// ---------------------------------------------------------------------------
// Host launcher
// ---------------------------------------------------------------------------
extern "C" void kernel_launch(void* const* d_in, const int* in_sizes, int n_in,
                              void* d_out, int out_size, void* d_ws, size_t ws_size,
                              hipStream_t stream) {
    (void)in_sizes; (void)n_in; (void)out_size; (void)ws_size;

    const float* hidden = (const float*)d_in[0];
    const float* amask  = (const float*)d_in[1];
    const float* relP   = (const float*)d_in[2];
    const float* rel2   = (const float*)d_in[3];
    const float* Wq     = (const float*)d_in[4];
    const float* bq     = (const float*)d_in[5];
    const float* Wk     = (const float*)d_in[6];
    const float* bk     = (const float*)d_in[7];
    const float* Wv     = (const float*)d_in[8];
    const float* bv     = (const float*)d_in[9];
    float* out = (float*)d_out;

    // Workspace layout (256B aligned)
    char* ws = (char*)d_ws;
    size_t off = 0;
    auto alloc = [&](size_t bytes) {
        void* p = ws + off;
        off = (off + bytes + 255) & ~(size_t)255;
        return p;
    };
    _Float16* X16  = (_Float16*)alloc((size_t)B_ * S_ * HID_ * sizeof(_Float16));
    _Float16* W16t = (_Float16*)alloc((size_t)NQKV * HID_ * sizeof(_Float16));
    float*    bQKV = (float*)   alloc((size_t)NQKV * sizeof(float));
    _Float16* Q16  = (_Float16*)alloc((size_t)B_ * NH_ * S_ * HD_ * sizeof(_Float16));
    _Float16* K16  = (_Float16*)alloc((size_t)B_ * NH_ * S_ * HD_ * sizeof(_Float16));
    _Float16* V16t = (_Float16*)alloc((size_t)B_ * NH_ * HD_ * S_ * sizeof(_Float16));

    // 1) Convert / pack
    {
        const int n = B_ * S_ * HID_;
        cvt_x_kernel<<<(n + 255) / 256, 256, 0, stream>>>(hidden, X16, n);
        const int nw = NQKV * HID_;
        pack_w_kernel<<<(nw + 255) / 256, 256, 0, stream>>>(Wq, Wk, Wv, W16t);
        pack_bias_kernel<<<(NQKV + 255) / 256, 256, 0, stream>>>(bq, bk, bv, bQKV);
    }

    // 2) Fused QKV projection GEMM
    {
        dim3 grid(NQKV / 16 / 4, (B_ * S_) / 16);          // (36, 256), 4 waves/block
        qkv_gemm_kernel<<<grid, 128, 0, stream>>>(X16, W16t, bQKV, Q16, K16, V16t);
    }

    // 3) Attention: 100 KB dynamic LDS (scores + probs + async bias staging)
    {
        dim3 grid(S_ / 16, B_ * NH_);                      // (64, 48)
        const size_t lds = SC_BYTES + PF_BYTES + REL_BYTES;
        attn_kernel<<<grid, 32, lds, stream>>>(Q16, K16, V16t, relP, rel2, amask, out);
    }
}
